// CorrelationPyramid_65773129171494
// MI455X (gfx1250) — matrix operand order repaired and checked by reading, not compiled, not run on hardware
//
#include <hip/hip_runtime.h>
#include <hip/hip_fp16.h>
#include <math.h>
#include <stdint.h>

// ---------------- problem constants (from reference setup_inputs) ----------
#define B_   2
#define N_   4
#define S_   5          // slot 0 = query, slots 1..4 = values
#define C_   128
#define CO_  64
#define H0_  160
#define W0_  160
#define HW0_ (H0_*W0_)

// padded LDS row layout for staged feature rows: 64 halves = 128B data,
// +16B pad per pixel => 144B stride (9x16B, keeps b128 alignment, rotates banks)
#define PADB  144
#define PADH  72       // stride in halves

// ---------------- CDNA5 WMMA types -----------------------------------------
typedef __attribute__((ext_vector_type(16))) _Float16     v16h;
typedef __attribute__((ext_vector_type(8)))  float        v8f;
typedef __attribute__((ext_vector_type(4)))  unsigned int u32x4;
typedef __attribute__((ext_vector_type(4)))  float        f32x4;

union V16H { v16h v; u32x4 u[2]; };
union V8F  { v8f  v; float f[8]; };
union U8H  { _Float16 h[8]; u32x4 u; };

__device__ __forceinline__ v8f wmma_f16(v16h a, v16h b, v8f c) {
  // D(f32 16x16) = A(f16 16x32) * B(f16 32x16) + C
  return __builtin_amdgcn_wmma_f32_16x16x32_f16(
      /*neg_a=*/false, a, /*neg_b=*/false, b,
      /*c_mod=*/(short)0, c, /*reuse_a=*/false, /*reuse_b=*/false);
}

// Async copy one contiguous global feature row (Wl*64 f16 = Wl*128 B) into a
// padded LDS row buffer using ASYNCcnt-tracked global_load_async_to_lds_b128.
// Each lane moves 16B per instruction; tracked by ASYNCcnt, not LOADcnt.
__device__ __forceinline__ void stage_row_async(
    unsigned sm_base,             // LDS byte offset of destination buffer
    const _Float16* srcRow,       // global row base
    int Wl, int t)                // t = linear thread id (0..127)
{
  const int chunks = Wl * 8;      // 8 x 16B chunks per pixel
  const unsigned long long base = (unsigned long long)(const void*)srcRow;
  for (int c = t; c < chunks; c += 128) {
    const unsigned pix = ((unsigned)c) >> 3;
    const unsigned sub = ((unsigned)c) & 7u;
    const unsigned lds = sm_base + pix * (unsigned)PADB + sub * 16u;
    const unsigned gof = ((unsigned)c) * 16u;
    asm volatile("global_load_async_to_lds_b128 %0, %1, %2"
                 :: "v"(lds), "v"(gof), "s"(base) : "memory");
  }
}

__device__ __forceinline__ void wait_async_all() {
  asm volatile("s_wait_asynccnt 0x0" ::: "memory");
}

// ===========================================================================
// Kernel 1: 1x1 conv projection (C=128 -> Co=64) via WMMA, f16 channel-last
// output + per-pixel inverse L2 norms. Block = 128 threads (4 waves),
// each block processes 8 strips of 16 pixels; wave w owns co-tile [16w,16w+16).
// ===========================================================================
__global__ __launch_bounds__(128) void proj_kernel(
    const float* __restrict__ qf,   // [B,C,H,W]
    const float* __restrict__ vf,   // [B,N,C,H,W]
    const float* __restrict__ pw,   // [Co,C]
    const float* __restrict__ pb,   // [Co]
    _Float16*    __restrict__ feat, // [B,S,HW,Co] f16 (unnormalized)
    float*       __restrict__ invn) // [B,S,HW]
{
  __shared__ alignas(16) _Float16 lds_w[CO_*C_];     // f16 weights, row-major
  __shared__ float                lds_b[CO_];
  __shared__ alignas(16) _Float16 lds_xt[16][C_+8];  // X tile transposed [pix][c]
  __shared__ float                norm_acc[16];

  const int t    = threadIdx.x;
  const int lane = t & 31;
  const int wv   = t >> 5;

  // stage weights (f32 -> f16) once per block
  #pragma unroll
  for (int i = 0; i < (CO_*C_)/128; ++i) {
    int idx = i*128 + t;
    lds_w[idx] = (_Float16)pw[idx];
  }
  if (t < CO_) lds_b[t] = pb[t];
  __syncthreads();

  const int lo  = lane & 15;
  const int hs  = (lane < 16) ? 0 : 8;   // accumulator M-offset
  const int ks  = (lane < 16) ? 0 : 16;  // B-frag K offset
  const int co0 = wv * 16;

  for (int it = 0; it < 8; ++it) {
    const int strip = blockIdx.x * 8 + it;          // 16-pixel strip id
    const int img   = strip / (HW0_/16);            // (b,s) image
    const int hw0   = (strip % (HW0_/16)) * 16;
    const int b     = img / S_;
    const int s     = img % S_;
    const float* src = (s == 0)
        ? (qf + (size_t)b * C_ * HW0_)
        : (vf + (size_t)(b*N_ + (s-1)) * C_ * HW0_);

    // stage X tile transposed: thread t = channel t, 16 pixels, f32->f16
    {
      const f32x4* s4 = (const f32x4*)(src + (size_t)t*HW0_ + hw0);
      f32x4 xv[4];
      #pragma unroll
      for (int jj = 0; jj < 4; ++jj) xv[jj] = s4[jj];
      #pragma unroll
      for (int j = 0; j < 16; ++j) lds_xt[j][t] = (_Float16)xv[j>>2][j&3];
    }
    if (t < 16) norm_acc[t] = 0.f;
    __syncthreads();

    // WMMA: D[co 16][pix 16] += W'[co][k] * X[k][pix], K = 128 in 4 steps
    v8f zero = {};
    V8F acc; acc.v = zero;
    #pragma unroll
    for (int k0 = 0; k0 < C_; k0 += 32) {
      V16H a, bx;
      const u32x4* ap = (const u32x4*)&lds_w[(co0+lo)*C_ + k0 + ks];
      a.u[0] = ap[0];            // K ks+0..7   -> vgpr 0..3
      a.u[1] = ap[1];            // K ks+8..15  -> vgpr 4..7
      const u32x4* bp = (const u32x4*)&lds_xt[lo][k0 + ks];
      bx.u[0] = bp[0];           // K ks+0..7
      bx.u[1] = bp[1];           // K ks+8..15
      acc.v = wmma_f16(a.v, bx.v, acc.v);
    }

    // epilogue: + bias, accumulate |y|^2 per pixel, store 8 f16 per lane
    float ss = 0.f;
    U8H pk;
    #pragma unroll
    for (int i = 0; i < 8; ++i) {
      float val = acc.f[i] + lds_b[co0 + hs + i];
      ss += val * val;
      pk.h[i] = (_Float16)val;
    }
    const size_t fo = (size_t)img*HW0_*CO_ + (size_t)(hw0+lo)*CO_ + co0 + hs;
    *(u32x4*)(feat + fo) = pk.u;               // 16B store, b128
    atomicAdd(&norm_acc[lo], ss);
    __syncthreads();
    if (t < 16) {
      float nrm = sqrtf(norm_acc[t]);
      invn[(size_t)img*HW0_ + hw0 + t] = 1.f / fmaxf(nrm, 1e-12f);
    }
    __syncthreads();
  }
}

// ===========================================================================
// Kernel 2: exact 2x2 average pool on f16 channel-last feats (+ inv norms).
// Pooling unnormalized feats by 2 repeatedly == reference's scale-s mean.
// ===========================================================================
__global__ __launch_bounds__(256) void pool_kernel(
    const _Float16* __restrict__ fin,  // [B,S,Win,Win,Co]
    _Float16*       __restrict__ fout, // [B,S,Wout,Wout,Co]
    float*          __restrict__ invo, // [B,S,Wout,Wout]
    int Win, int Wout)
{
  const int total = B_*S_*Wout*Wout;
  const int idx = blockIdx.x * blockDim.x + threadIdx.x;
  if (idx >= total) return;
  const int img = idx / (Wout*Wout);
  const int rem = idx % (Wout*Wout);
  const int oh = rem / Wout, ow = rem % Wout;
  const size_t ib = ((size_t)img*Win*Win + (size_t)(2*oh)*Win + 2*ow) * CO_;
  const u32x4* r00 = (const u32x4*)(fin + ib);
  const u32x4* r01 = (const u32x4*)(fin + ib + CO_);
  const u32x4* r10 = (const u32x4*)(fin + ib + (size_t)Win*CO_);
  const u32x4* r11 = (const u32x4*)(fin + ib + (size_t)Win*CO_ + CO_);
  u32x4* dst = (u32x4*)(fout + (size_t)idx*CO_);

  float ss = 0.f;
  #pragma unroll
  for (int j = 0; j < CO_/8; ++j) {
    U8H a, b, c, d, o;
    a.u = r00[j]; b.u = r01[j]; c.u = r10[j]; d.u = r11[j];
    #pragma unroll
    for (int i = 0; i < 8; ++i) {
      float m = 0.25f * ((float)a.h[i] + (float)b.h[i] +
                         (float)c.h[i] + (float)d.h[i]);
      ss += m * m;
      o.h[i] = (_Float16)m;
    }
    dst[j] = o.u;
  }
  invo[idx] = 1.f / fmaxf(sqrtf(ss), 1e-12f);
}

// ===========================================================================
// Kernel 3: banded-Gram correlation via WMMA with async-to-LDS row staging.
// Block (4 waves) owns one (b,n,h): q row staged once, v rows double-buffered
// with global_load_async_to_lds_b128 prefetch across the dy loop (ASYNCcnt).
// One wave-tile: D[m][j] = q[w0+m] . v[(w0-r+j) mod W]; j-tiles {0..15,16..31}
// cover the +-r band; K=64 in two f16 WMMA steps.  jnp.roll semantics:
// v index is ((h-dy) mod H, (w-dx) mod W)  =>  j = m + 2r - dxIdx.
// out = D * invq * invv  implements x/max(||x||,1e-12) normalization.
// ===========================================================================
__global__ __launch_bounds__(128) void corr_row_kernel(
    const _Float16* __restrict__ feat, const float* __restrict__ invn,
    float* __restrict__ out, int Wl, int r, int ntW)
{
  extern __shared__ _Float16 smem[];   // [Q row | V row buf0 | V row buf1], PADB stride
  const int Hl = Wl;                   // square levels
  const int nd = 2*r + 1;
  const int t    = threadIdx.x;
  const int lane = t & 31;
  const int wv   = t >> 5;

  int rid = blockIdx.x;
  const int h = rid % Hl; rid /= Hl;
  const int n = rid % N_;
  const int b = rid / N_;

  const size_t qoff = (size_t)(b*S_) * Hl*Wl + (size_t)h * Wl;
  const _Float16* qrowG = feat + qoff * CO_;
  const float*    invq  = invn + qoff;
  const size_t vpo = (size_t)(b*S_ + 1 + n) * Hl*Wl;
  const _Float16* vplane = feat + vpo * CO_;
  const float*    invvP  = invn + vpo;

  // LDS byte offsets (generic->LDS: low 32 bits of the flat address)
  const unsigned smq = (unsigned)(uintptr_t)(void*)smem;
  const unsigned smv0 = smq + (unsigned)Wl * PADB;
  const unsigned smv1 = smq + 2u * (unsigned)Wl * PADB;
  _Float16* smqP = smem;
  _Float16* smvP[2] = { smem + (size_t)Wl*PADH, smem + 2*(size_t)Wl*PADH };
  const unsigned smvOff[2] = { smv0, smv1 };

  auto hpOf = [&](int dyIdx) {
    int hp = h - (dyIdx - r); hp %= Hl; if (hp < 0) hp += Hl; return hp;
  };

  // initial stage: q row + v row for dyI=0
  stage_row_async(smq,    qrowG, Wl, t);
  stage_row_async(smv0,   vplane + (size_t)hpOf(0)*Wl*CO_, Wl, t);
  wait_async_all();
  __syncthreads();

  const int lo = lane & 15;
  const int hs = (lane < 16) ? 0 : 8;
  const int ks = (lane < 16) ? 0 : 16;
  const int nsh = nd * nd;
  const size_t obase = (((size_t)(b*N_ + n)*Hl + h) * (size_t)Wl) * nsh;

  for (int dyI = 0; dyI < nd; ++dyI) {
    const int cur = dyI & 1;
    // prefetch next v row into the other buffer while computing this one
    if (dyI + 1 < nd)
      stage_row_async(smvOff[cur ^ 1], vplane + (size_t)hpOf(dyI+1)*Wl*CO_, Wl, t);

    const _Float16* sv = smvP[cur];
    const float* invv = invvP + (size_t)hpOf(dyI) * Wl;

    for (int wt = wv; wt < ntW; wt += 4) {
      const int w0 = wt * 16;
      int qpix = w0 + lo; if (qpix >= Wl) qpix = Wl - 1;   // writes guarded
      int c0 = w0 - r + lo;      c0 = ((c0 % Wl) + Wl) % Wl;
      int c1 = w0 - r + 16 + lo; c1 = ((c1 % Wl) + Wl) % Wl;

      v8f zero = {};
      V8F acc0, acc1; acc0.v = zero; acc1.v = zero;
      #pragma unroll
      for (int k0 = 0; k0 < CO_; k0 += 32) {
        V16H a, b0, b1;
        const u32x4* ap  = (const u32x4*)(smqP + (size_t)qpix*PADH + k0 + ks);
        a.u[0] = ap[0]; a.u[1] = ap[1];
        const u32x4* bp0 = (const u32x4*)(sv + (size_t)c0*PADH + k0 + ks);
        b0.u[0] = bp0[0]; b0.u[1] = bp0[1];
        const u32x4* bp1 = (const u32x4*)(sv + (size_t)c1*PADH + k0 + ks);
        b1.u[0] = bp1[0]; b1.u[1] = bp1[1];
        acc0.v = wmma_f16(a.v, b0.v, acc0.v);
        acc1.v = wmma_f16(a.v, b1.v, acc1.v);
      }

      const float iv0 = invv[c0], iv1 = invv[c1];
      #pragma unroll
      for (int i = 0; i < 8; ++i) {
        const int m = hs + i;
        const int wpix = w0 + m;
        if (wpix >= Wl) continue;
        const float qi = invq[wpix];
        const size_t orow = obase + (size_t)wpix*nsh + (size_t)dyI*nd;
        const int dx0 = m + 2*r - lo;        // j = lo
        if (dx0 >= 0 && dx0 < nd) out[orow + dx0] = acc0.f[i] * qi * iv0;
        const int dx1 = m + 2*r - 16 - lo;   // j = 16 + lo
        if (dx1 >= 0 && dx1 < nd) out[orow + dx1] = acc1.f[i] * qi * iv1;
      }
    }

    // ensure our prefetch landed, then block-wide visibility for next dy
    wait_async_all();
    __syncthreads();
  }
}

// ===========================================================================
extern "C" void kernel_launch(void* const* d_in, const int* in_sizes, int n_in,
                              void* d_out, int out_size, void* d_ws, size_t ws_size,
                              hipStream_t stream) {
  (void)in_sizes; (void)n_in; (void)out_size; (void)ws_size;
  const float* qf = (const float*)d_in[0];
  const float* vf = (const float*)d_in[1];
  const float* pw = (const float*)d_in[2];
  const float* pb = (const float*)d_in[3];
  float* out = (float*)d_out;

  // workspace layout: f16 feats per level (channel-last), then f32 inv-norms
  constexpr size_t PIX0 = (size_t)B_*S_*160*160;   // 256000
  constexpr size_t PIX1 = (size_t)B_*S_*80*80;     //  64000
  constexpr size_t PIX2 = (size_t)B_*S_*40*40;     //  16000
  constexpr size_t PIX3 = (size_t)B_*S_*20*20;     //   4000
  constexpr size_t F0 = 0;
  constexpr size_t F1 = F0 + PIX0*CO_;
  constexpr size_t F2 = F1 + PIX1*CO_;
  constexpr size_t F3 = F2 + PIX2*CO_;
  constexpr size_t FTOT = F3 + PIX3*CO_;           // ~43.5 MB of f16 (L2-resident)
  constexpr size_t I0 = 0, I1 = PIX0, I2 = I1+PIX1, I3 = I2+PIX2;

  _Float16* feat = (_Float16*)d_ws;
  float*    invn = (float*)(feat + FTOT);

  // output offsets (floats), concatenated tuple in return order
  constexpr size_t O0 = 0;
  constexpr size_t O1 = O0 + (size_t)B_*N_*160*160*81;
  constexpr size_t O2 = O1 + (size_t)B_*N_*80*80*25;
  constexpr size_t O3 = O2 + (size_t)B_*N_*40*40*9;

  // 1) projection + level-0 inv norms   (16000 strips / 8 per block)
  proj_kernel<<<dim3((unsigned)(PIX0/(16*8))), dim3(128), 0, stream>>>(
      qf, vf, pw, pb, feat + F0, invn + I0);

  // 2) pyramid pooling (exact 2x2 means compose to scale 2/4/8)
  pool_kernel<<<dim3((unsigned)((PIX1+255)/256)), dim3(256), 0, stream>>>(
      feat + F0, feat + F1, invn + I1, 160, 80);
  pool_kernel<<<dim3((unsigned)((PIX2+255)/256)), dim3(256), 0, stream>>>(
      feat + F1, feat + F2, invn + I2, 80, 40);
  pool_kernel<<<dim3((unsigned)((PIX3+255)/256)), dim3(256), 0, stream>>>(
      feat + F2, feat + F3, invn + I3, 40, 20);

  // 3) correlations: one block per (b,n,h); LDS = q row + 2 v row buffers
  {
    const int Wl = 160, r = 4, ntW = 10;
    const unsigned grid = (unsigned)(B_*N_*Wl);
    const size_t shmem = 3u * (size_t)Wl * PADB;   // 69120 B
    corr_row_kernel<<<dim3(grid), dim3(128), shmem, stream>>>(
        feat + F0, invn + I0, out + O0, Wl, r, ntW);
  }
  {
    const int Wl = 80, r = 2, ntW = 5;
    const unsigned grid = (unsigned)(B_*N_*Wl);
    const size_t shmem = 3u * (size_t)Wl * PADB;
    corr_row_kernel<<<dim3(grid), dim3(128), shmem, stream>>>(
        feat + F1, invn + I1, out + O1, Wl, r, ntW);
  }
  {
    const int Wl = 40, r = 1, ntW = 3;
    const unsigned grid = (unsigned)(B_*N_*Wl);
    const size_t shmem = 3u * (size_t)Wl * PADB;
    corr_row_kernel<<<dim3(grid), dim3(128), shmem, stream>>>(
        feat + F2, invn + I2, out + O2, Wl, r, ntW);
  }
  {
    const int Wl = 20, r = 1, ntW = 2;
    const unsigned grid = (unsigned)(B_*N_*Wl);
    const size_t shmem = 3u * (size_t)Wl * PADB;
    corr_row_kernel<<<dim3(grid), dim3(128), shmem, stream>>>(
        feat + F3, invn + I3, out + O3, Wl, r, ntW);
  }
}
